// Mamba3SISOBlock_57518202028648
// MI455X (gfx1250) — matrix operand order repaired and checked by reading, not compiled, not run on hardware
//
#include <hip/hip_runtime.h>
#include <hip/hip_bf16.h>
#include <math.h>

// ---------------- problem constants ----------------
#define BATCHN   2
#define SEQLEN   2048
#define DMODEL   1024
#define DINNER   2048
#define HEADDIM  64
#define NHEADS   32
#define DSTATE   128
#define NANG     32
#define PDIM     4480           // IN_PROJ_DIM
#define NTOK     (BATCHN*SEQLEN) // 4096
// proj offsets
#define OFF_Z    0
#define OFF_V    2048
#define OFF_B    4096
#define OFF_C    4224
#define OFF_DT   4352
#define OFF_A    4384
#define OFF_TRAP 4416
#define OFF_ANG  4448

typedef __bf16 bf16;
typedef __bf16 v16bf __attribute__((ext_vector_type(16)));
typedef __bf16 v8bf  __attribute__((ext_vector_type(8)));
typedef float  v8f   __attribute__((ext_vector_type(8)));

union ABFrag { v16bf v; struct { v8bf lo, hi; } h; };

__device__ __forceinline__ bf16 to_bf16(float f) {
    union { float f; unsigned u; } a; a.f = f;
    unsigned r = a.u + 0x7FFFu + ((a.u >> 16) & 1u);   // round-to-nearest-even
    union { unsigned short s; bf16 b; } o; o.s = (unsigned short)(r >> 16);
    return o.b;
}

// ---------------- fp32 -> bf16 conversion ----------------
__global__ void cvt_bf16_kernel(const float* __restrict__ in, bf16* __restrict__ out, size_t n) {
    size_t i = (size_t)blockIdx.x * blockDim.x + threadIdx.x;
    size_t stride = (size_t)gridDim.x * blockDim.x;
    for (; i < n; i += stride) out[i] = to_bf16(in[i]);
}

// ---------------- bf16 WMMA GEMM: C[M,N] = A[M,K] * B[N,K]^T ----------------
// A row-major (lda=K), B stored [n][k] k-contiguous (ldb=K), C row-major (ldc=N).
__device__ __forceinline__ v16bf load_a_frag(const bf16* A, int lda, int m0, int k0, int lane) {
    int row = m0 + (lane & 15);
    int kb  = k0 + (lane >> 4) * 8;           // ISA 16-bit A layout: lanes>=16 start at K+8
    const bf16* p = A + (size_t)row * lda + kb;
    ABFrag u;
    u.h.lo = *(const v8bf*)p;                  // K kb..kb+7   -> VGPR0..3
    u.h.hi = *(const v8bf*)(p + 16);           // K kb+16..+23 -> VGPR4..7
    return u.v;
}
__device__ __forceinline__ v16bf load_b_frag(const bf16* B, int ldb, int n0, int k0, int lane) {
    int col = n0 + (lane & 15);
    int kb  = k0 + (lane >> 4) * 8;           // same K layout as A: lanes>=16 start at K+8
    const bf16* p = B + (size_t)col * ldb + kb;
    ABFrag u;
    u.h.lo = *(const v8bf*)p;
    u.h.hi = *(const v8bf*)(p + 16);
    return u.v;
}
__device__ __forceinline__ void store_c(float* C, int ldc, int m0, int n0, int lane, v8f c) {
    int col   = n0 + (lane & 15);
    int rbase = m0 + (lane >> 4) * 8;         // ISA C/D layout
    #pragma unroll
    for (int r = 0; r < 8; ++r) C[(size_t)(rbase + r) * ldc + col] = c[r];
}

__global__ __launch_bounds__(256) void gemm_bf16_kernel(
    const bf16* __restrict__ A, const bf16* __restrict__ B, float* __restrict__ C,
    int K, int N)
{
    const int lane = threadIdx.x & 31;
    const int wave = threadIdx.x >> 5;        // 8 waves
    const int wm = wave & 3, wn = wave >> 2;  // 4x2 wave grid
    const int m0 = blockIdx.x * 128 + wm * 32;
    const int n0 = blockIdx.y * 64  + wn * 32;

    v8f c00 = {}, c01 = {}, c10 = {}, c11 = {};
    for (int k0 = 0; k0 < K; k0 += 32) {
        v16bf a0 = load_a_frag(A, K, m0,      k0, lane);
        v16bf a1 = load_a_frag(A, K, m0 + 16, k0, lane);
        v16bf b0 = load_b_frag(B, K, n0,      k0, lane);
        v16bf b1 = load_b_frag(B, K, n0 + 16, k0, lane);
        c00 = __builtin_amdgcn_wmma_f32_16x16x32_bf16(false, a0, false, b0, (short)0, c00, false, false);
        c01 = __builtin_amdgcn_wmma_f32_16x16x32_bf16(false, a0, false, b1, (short)0, c01, false, false);
        c10 = __builtin_amdgcn_wmma_f32_16x16x32_bf16(false, a1, false, b0, (short)0, c10, false, false);
        c11 = __builtin_amdgcn_wmma_f32_16x16x32_bf16(false, a1, false, b1, (short)0, c11, false, false);
    }
    store_c(C, N, m0,      n0,      lane, c00);
    store_c(C, N, m0,      n0 + 16, lane, c01);
    store_c(C, N, m0 + 16, n0,      lane, c10);
    store_c(C, N, m0 + 16, n0 + 16, lane, c11);
}

// ---------------- sequential SSM scan ----------------
// grid = BATCH * NHEADS * PSPLIT blocks; each block owns PROWS of the 64 p-rows
// for one (b,h) and walks t = 0..SEQLEN-1 sequentially, carrying state/pk/pv in VGPRs.
#define PSPLIT 4
#define PROWS  (HEADDIM / PSPLIT)   // 16
#define NGRP   (256 / PROWS)        // 16
#define NCH    (DSTATE / NGRP)      // 8

__device__ __forceinline__ float softplusf(float x) { return x > 20.f ? x : log1pf(expf(x)); }

__global__ __launch_bounds__(256) void scan_kernel(
    const float* __restrict__ proj, const float* __restrict__ dt_bias,
    const float* __restrict__ Bnw,  const float* __restrict__ Cnw,
    const float* __restrict__ Bb,   const float* __restrict__ Cb,
    const float* __restrict__ Dp,   float* __restrict__ y)
{
    const int blk = blockIdx.x;
    const int ps  = blk % PSPLIT;
    const int h   = (blk / PSPLIT) % NHEADS;
    const int bat = blk / (PSPLIT * NHEADS);
    const int t   = threadIdx.x;
    const int p_local = t % PROWS;
    const int p   = ps * PROWS + p_local;
    const int n0  = (t / PROWS) * NCH;

    __shared__ float sB[DSTATE], sC[DSTATE], red[256], ph[NANG], sc[4];
    if (t < NANG) ph[t] = 0.f;

    float st[NCH], pk[NCH];
    #pragma unroll
    for (int j = 0; j < NCH; ++j) { st[j] = 0.f; pk[j] = 0.f; }
    float pv  = 0.f;
    const float dpH = Dp[h];
    const float dtb = dt_bias[h];
    const size_t tokbase = (size_t)bat * SEQLEN;

    for (int s = 0; s < SEQLEN; ++s) {
        __syncthreads();  // protect sB/sC/ph/red from previous iteration's readers
        const float* pr = proj + (tokbase + s) * (size_t)PDIM;
        if (t < DSTATE) sB[t] = pr[OFF_B + t];
        else            sC[t - DSTATE] = pr[OFF_C + (t - DSTATE)];
        if (t < NANG)   ph[t] += pr[OFF_ANG + t];            // phase cumsum
        if (t == 0) {
            float dt    = softplusf(pr[OFF_DT + h] + dtb);
            float Av    = fminf(-softplusf(pr[OFF_A + h]), -1e-4f);
            float trap  = 1.f / (1.f + expf(-pr[OFF_TRAP + h]));
            float alpha = expf(Av * dt);
            sc[0] = alpha; sc[1] = (1.f - trap) * dt * alpha; sc[2] = trap * dt;
        }
        __syncthreads();
        // rmsnorm reductions for B and C (sum of squares)
        float q = (t < DSTATE) ? sB[t] * sB[t] : sC[t - DSTATE] * sC[t - DSTATE];
        red[t] = q;
        __syncthreads();
        for (int k = 64; k >= 1; k >>= 1) {
            if (t < k) red[t] += red[t + k];
            else if (t >= 128 && t < 128 + k) red[t] += red[t + k];
            __syncthreads();
        }
        const float rB = sqrtf(red[0]   * (1.f / DSTATE) + 1e-5f);
        const float rC = sqrtf(red[128] * (1.f / DSTATE) + 1e-5f);
        if (t < DSTATE) sB[t] = sB[t] / rB * Bnw[t] + Bb[h * DSTATE + t];
        else { int n = t - DSTATE; sC[n] = sC[n] / rC * Cnw[n] + Cb[h * DSTATE + n]; }
        __syncthreads();
        // rotary on first 64 dims (32 pairs each of B and C)
        if (t < 2 * NANG) {
            float* arr = (t < NANG) ? sB : sC;
            int i = t & (NANG - 1);
            float cs = cosf(ph[i]), sn = sinf(ph[i]);
            float x0 = arr[2 * i], x1 = arr[2 * i + 1];
            arr[2 * i]     = x0 * cs - x1 * sn;
            arr[2 * i + 1] = x0 * sn + x1 * cs;
        }
        __syncthreads();
        const float alpha = sc[0], beta = sc[1], gamma = sc[2];
        const float v = pr[OFF_V + h * HEADDIM + p];
        float acc = 0.f;
        #pragma unroll
        for (int j = 0; j < NCH; ++j) {
            int n = n0 + j;
            float kn  = sB[n];
            float stv = alpha * st[j] + beta * (pv * pk[j]) + gamma * (v * kn);
            st[j] = stv;
            acc  += stv * sC[n];
            pk[j] = kn;
        }
        pv = v;
        red[t] = acc;
        __syncthreads();
        if (t < PROWS) {   // t == p_local here
            float ys = dpH * v;
            #pragma unroll
            for (int g = 0; g < NGRP; ++g) ys += red[g * PROWS + t];
            y[(tokbase + s) * (size_t)DINNER + h * HEADDIM + p] = ys;
        }
    }
}

// ---------------- gated output RMS norm, emits bf16 for the out-proj GEMM ----------------
__global__ __launch_bounds__(256) void gate_norm_kernel(
    const float* __restrict__ y, const float* __restrict__ proj,
    const float* __restrict__ onw, bf16* __restrict__ ybf)
{
    const int tok = blockIdx.x;
    const float* yr = y    + (size_t)tok * DINNER;
    const float* zr = proj + (size_t)tok * PDIM;   // z at offset 0
    __shared__ float red[256];
    float s = 0.f;
    for (int i = threadIdx.x; i < DINNER; i += 256) { float v = yr[i]; s += v * v; }
    red[threadIdx.x] = s;
    __syncthreads();
    for (int k = 128; k >= 1; k >>= 1) {
        if (threadIdx.x < k) red[threadIdx.x] += red[threadIdx.x + k];
        __syncthreads();
    }
    const float rms = sqrtf(red[0] * (1.f / DINNER) + 1e-5f);
    for (int i = threadIdx.x; i < DINNER; i += 256) {
        float z = zr[i];
        float g = z / (1.f + expf(-z));            // silu
        ybf[(size_t)tok * DINNER + i] = to_bf16(yr[i] / rms * onw[i] * g);
    }
}

// ---------------- launcher ----------------
extern "C" void kernel_launch(void* const* d_in, const int* in_sizes, int n_in,
                              void* d_out, int out_size, void* d_ws, size_t ws_size,
                              hipStream_t stream) {
    (void)in_sizes; (void)n_in; (void)out_size; (void)ws_size;
    const float* x       = (const float*)d_in[0];
    const float* Win     = (const float*)d_in[1];
    const float* dt_bias = (const float*)d_in[2];
    const float* Bnw     = (const float*)d_in[3];
    const float* Cnw     = (const float*)d_in[4];
    const float* Bb      = (const float*)d_in[5];
    const float* Cb      = (const float*)d_in[6];
    const float* Dp      = (const float*)d_in[7];
    const float* onw     = (const float*)d_in[8];
    const float* Wout    = (const float*)d_in[9];
    float* out = (float*)d_out;

    char* w = (char*)d_ws;
    size_t o = 0;
    auto carve = [&](size_t bytes) -> void* {
        void* p = w + o; o = (o + bytes + 255) & ~(size_t)255; return p;
    };
    float* proj = (float*)carve((size_t)NTOK * PDIM   * 4);  // 73.4 MB
    float* yraw = (float*)carve((size_t)NTOK * DINNER * 4);  // 33.6 MB
    bf16*  xbf  = (bf16*) carve((size_t)NTOK * DMODEL * 2);  //  8.4 MB
    bf16*  wibf = (bf16*) carve((size_t)PDIM * DMODEL * 2);  //  9.2 MB
    bf16*  ybf  = (bf16*) carve((size_t)NTOK * DINNER * 2);  // 16.8 MB
    bf16*  wobf = (bf16*) carve((size_t)DMODEL * DINNER * 2);//  4.2 MB

    cvt_bf16_kernel<<<2048, 256, 0, stream>>>(x,    xbf,  (size_t)NTOK * DMODEL);
    cvt_bf16_kernel<<<2048, 256, 0, stream>>>(Win,  wibf, (size_t)PDIM * DMODEL);
    cvt_bf16_kernel<<<2048, 256, 0, stream>>>(Wout, wobf, (size_t)DMODEL * DINNER);

    // in-proj: (4096 x 1024) * (4480 x 1024)^T -> proj (4096 x 4480)
    gemm_bf16_kernel<<<dim3(NTOK / 128, PDIM / 64), 256, 0, stream>>>(xbf, wibf, proj, DMODEL, PDIM);

    // sequential scan over the sequence
    scan_kernel<<<BATCHN * NHEADS * PSPLIT, 256, 0, stream>>>(
        proj, dt_bias, Bnw, Cnw, Bb, Cb, Dp, yraw);

    // gated RMS norm -> bf16
    gate_norm_kernel<<<NTOK, 256, 0, stream>>>(yraw, proj, onw, ybf);

    // out-proj: (4096 x 2048) * (1024 x 2048)^T -> out (4096 x 1024)
    gemm_bf16_kernel<<<dim3(NTOK / 128, DMODEL / 64), 256, 0, stream>>>(ybf, wobf, out, DINNER, DMODEL);
}
